// FactorizedSignEncoder_71047349010698
// MI455X (gfx1250) — compile-verified, matrix operand order checked by reading, not run on hardware
//
#include <hip/hip_runtime.h>
#include <hip/hip_bf16.h>
#include <stdint.h>

// ---------------------------------------------------------------------------
// CDNA5 (gfx1250) WMMA + TDM implementation of the factorized VQ encoder.
//   - all GEMM operands pre-converted to bf16 (one-shot conversion kernels)
//   - tiles staged into LDS by tensor_load_to_lds (TDM); descriptor padding
//     produces the 40-element LDS row stride; DOUBLE-BUFFERED so the TDM
//     engine streams tile t+1 while the SIMDs run WMMA on tile t
//   - v_wmma_f32_16x16x32_bf16, f32 accumulate; LN / residual kept in f32
//   - fused codebook argmin via ordered-key atomicMin
// ---------------------------------------------------------------------------

typedef __attribute__((ext_vector_type(16))) __bf16 v16bf;
typedef __attribute__((ext_vector_type(8)))  float  v8f;
typedef __attribute__((ext_vector_type(4)))  unsigned int u32x4;
typedef __attribute__((ext_vector_type(8)))  int          i32x8;
typedef __attribute__((ext_vector_type(4)))  int          i32x4;

union FragBf { v16bf v; uint4 q[2]; };

#define M_TOK 16384   // B*T
#define EMB   256

#if defined(__HIP_DEVICE_COMPILE__) && __has_builtin(__builtin_amdgcn_tensor_load_to_lds)
#define USE_TDM 1
#else
#define USE_TDM 0
#endif

__device__ __forceinline__ __bf16 to_bf16(float f) { return (__bf16)f; }

__device__ __forceinline__ float gelu_tanh(float x) {
  const float c = 0.7978845608028654f;       // sqrt(2/pi)
  float x3 = x * x * x;
  return 0.5f * x * (1.0f + tanhf(c * (x + 0.044715f * x3)));
}

__device__ __forceinline__ unsigned ford(float f) {
  unsigned u = __float_as_uint(f);
  return (u & 0x80000000u) ? ~u : (u | 0x80000000u);   // monotone float->u32
}

#if USE_TDM
// ---------------------------------------------------------------------------
// TDM: DMA a 2D tile [rows x row_elems] of bf16 from global into LDS.
// pad_interval=16 DWORDs (one 64B row), pad_amount=4 DWORDs (8 elems)
//  => LDS row stride = 40 bf16 elements, matching the fragment loader.
// valid_elems: columns readable from tile start (OOB -> zero fill).
// ---------------------------------------------------------------------------
__device__ __forceinline__ void tdm_load_2d(unsigned lds_off, const void* gtile,
                                            unsigned rows, unsigned row_elems,
                                            unsigned valid_elems,
                                            unsigned stride_elems)
{
  unsigned long long ga = (unsigned long long)(uintptr_t)gtile;
  u32x4 g0;
  g0.x = 1u;                                             // count=1 (valid user D#)
  g0.y = lds_off;                                        // lds_addr (bytes)
  g0.z = (unsigned)ga;                                   // global_addr[31:0]
  g0.w = (unsigned)((ga >> 32) & 0x01FFFFFFu) | (2u << 30); // addr[56:32] | type=2
  i32x8 g1;
  g1[0] = (int)((1u << 16)            // data_size = 2 bytes
              | (1u << 20)            // pad_enable
              | (3u << 22)            // pad_interval: 16 DWORDs
              | (3u << 25));          // pad_amount : 4 DWORDs
  g1[1] = (int)((valid_elems & 0xffffu) << 16);                        // dim0 lo
  g1[2] = (int)(((valid_elems >> 16) & 0xffffu) | ((rows & 0xffffu) << 16)); // dim0 hi|dim1 lo
  g1[3] = (int)(((rows >> 16) & 0xffffu) | ((row_elems & 0xffffu) << 16));   // dim1 hi|tile_dim0
  g1[4] = (int)(rows & 0xffffu);                                       // tile_dim1 (dim2=0)
  g1[5] = (int)stride_elems;                                           // dim0_stride lo
  g1[6] = 0;
  g1[7] = 0;
  i32x4 z4 = {0, 0, 0, 0};
  i32x8 z8 = {0, 0, 0, 0, 0, 0, 0, 0};
  __builtin_amdgcn_tensor_load_to_lds(g0, g1, z4, z4, z8, 0);
}
#endif

// ---------------------------------------------------------------------------
// GEMM: C[M,N] = epi(A[M,K](bf16) @ Bt[N,K](bf16)^T + bias[N])
// EPI: 0 = bias, 1 = bias+GELU, 2 = bias+residual(f32)
// WF: write f32 output Cf;  WB: write bf16 output Cb
// block 128x128, wave 32x64 (2x4 WMMA 16x16x32 tiles), BK=32,
// double-buffered TDM staging (DMA of tile t+1 overlaps WMMA of tile t).
// ---------------------------------------------------------------------------
#define TILE_ELEMS (128 * 40)
#define TILE_BYTES (TILE_ELEMS * 2)

template <int EPI, bool WF, bool WB>
__global__ __launch_bounds__(256)
void gemm_bf16t(const __bf16* __restrict__ A, const __bf16* __restrict__ Bt,
                const float* __restrict__ bias, const float* __restrict__ Res,
                float* __restrict__ Cf, __bf16* __restrict__ Cb,
                int M, int N, int K)
{
  __shared__ __bf16 As[2][TILE_ELEMS];
  __shared__ __bf16 Bs[2][TILE_ELEMS];

  const int tid  = threadIdx.x;
  const int lane = tid & 31;
  const int wave = tid >> 5;
  const int wm   = (wave >> 1) * 32;   // 4 wave-rows  -> 128 M
  const int wn   = (wave & 1) * 64;    // 2 wave-cols  -> 128 N
  const int hi16 = lane >> 4;
  const int mr   = lane & 15;
  const int kb   = hi16 * 8;
  const int m0   = blockIdx.y * 128;
  const int n0   = blockIdx.x * 128;

  v8f c[2][4] = {};

  auto stage = [&](int buf, int k0) {
#if USE_TDM
    if (wave == 0)
      tdm_load_2d((unsigned)(uintptr_t)&As[buf][0], A + (size_t)m0 * K + k0,
                  128u, 32u, (unsigned)(K - k0), (unsigned)K);
    else if (wave == 1)
      tdm_load_2d((unsigned)(uintptr_t)&Bs[buf][0], Bt + (size_t)n0 * K + k0,
                  128u, 32u, (unsigned)(K - k0), (unsigned)K);
#else
    const int r = tid >> 1, cg = (tid & 1) * 16;
    {
      const __bf16* src = A + (size_t)(m0 + r) * K + k0 + cg;
      uint4 a0 = {0, 0, 0, 0}, a1 = {0, 0, 0, 0};
      if (k0 + cg < K) { a0 = ((const uint4*)src)[0]; a1 = ((const uint4*)src)[1]; }
      *(uint4*)&As[buf][r * 40 + cg]     = a0;
      *(uint4*)&As[buf][r * 40 + cg + 8] = a1;
    }
    {
      const __bf16* src = Bt + (size_t)(n0 + r) * K + k0 + cg;
      uint4 b0 = {0, 0, 0, 0}, b1 = {0, 0, 0, 0};
      if (k0 + cg < K) { b0 = ((const uint4*)src)[0]; b1 = ((const uint4*)src)[1]; }
      *(uint4*)&Bs[buf][r * 40 + cg]     = b0;
      *(uint4*)&Bs[buf][r * 40 + cg + 8] = b1;
    }
#endif
  };

  const int nk = (K + 31) / 32;
  stage(0, 0);

  for (int t = 0; t < nk; ++t) {
    const int buf = t & 1;
    const bool haveNext = (t + 1 < nk);
    if (haveNext) stage(buf ^ 1, (t + 1) * 32);
#if USE_TDM
    if (wave < 2) {
      if (haveNext) __builtin_amdgcn_s_wait_tensorcnt(1);  // tile t landed
      else          __builtin_amdgcn_s_wait_tensorcnt(0);  // drain
    }
#endif
    __syncthreads();

    FragBf a[2], b[4];
#pragma unroll
    for (int i = 0; i < 2; ++i) {
      const __bf16* p = &As[buf][(wm + 16 * i + mr) * 40 + kb];
      a[i].q[0] = *(const uint4*)p;
      a[i].q[1] = *(const uint4*)(p + 16);
    }
#pragma unroll
    for (int j = 0; j < 4; ++j) {
      const __bf16* p = &Bs[buf][(wn + 16 * j + mr) * 40 + kb];
      b[j].q[0] = *(const uint4*)p;
      b[j].q[1] = *(const uint4*)(p + 16);
    }
#pragma unroll
    for (int i = 0; i < 2; ++i)
#pragma unroll
      for (int j = 0; j < 4; ++j)
        c[i][j] = __builtin_amdgcn_wmma_f32_16x16x32_bf16(
            false, a[i].v, false, b[j].v, (short)0, c[i][j], false, false);
    __syncthreads();
  }

#pragma unroll
  for (int i = 0; i < 2; ++i) {
    const int row_base = m0 + wm + 16 * i + hi16 * 8;
#pragma unroll
    for (int j = 0; j < 4; ++j) {
      const int col = n0 + wn + 16 * j + mr;
      const float bv = bias[col];
#pragma unroll
      for (int r = 0; r < 8; ++r) {
        const size_t idx = (size_t)(row_base + r) * N + col;
        float v = c[i][j][r] + bv;
        if (EPI == 1) v = gelu_tanh(v);
        if (EPI == 2) v += Res[idx];
        if (WF) Cf[idx] = v;
        if (WB) Cb[idx] = to_bf16(v);
      }
    }
  }
}

// ---------------------------------------------------------------------------
// Row LayerNorm (+optional GELU); emits f32 (residual path) and bf16 (GEMM A).
// ---------------------------------------------------------------------------
template <int W, bool GELU_OUT>
__global__ __launch_bounds__(256)
void layernorm_k(const float* __restrict__ X, const float* __restrict__ g,
                 const float* __restrict__ b, float* __restrict__ Yf,
                 __bf16* __restrict__ Yb)
{
  constexpr int PT = W / 256;
  const int tid = threadIdx.x;
  const int row = blockIdx.x;
  const float* x = X + (size_t)row * W;
  float v[PT];
  float s = 0.f, s2 = 0.f;
#pragma unroll
  for (int i = 0; i < PT; ++i) {
    v[i] = x[tid + i * 256];
    s += v[i]; s2 += v[i] * v[i];
  }
  __shared__ float ss[256], sq[256];
  ss[tid] = s; sq[tid] = s2;
  __syncthreads();
  for (int off = 128; off > 0; off >>= 1) {
    if (tid < off) { ss[tid] += ss[tid + off]; sq[tid] += sq[tid + off]; }
    __syncthreads();
  }
  const float mean = ss[0] * (1.0f / W);
  const float var  = sq[0] * (1.0f / W) - mean * mean;
  const float rstd = rsqrtf(var + 1e-5f);
#pragma unroll
  for (int i = 0; i < PT; ++i) {
    const int cix = tid + i * 256;
    float o = (v[i] - mean) * rstd * g[cix] + b[cix];
    if (GELU_OUT) o = gelu_tanh(o);
    Yf[(size_t)row * W + cix] = o;
    Yb[(size_t)row * W + cix] = to_bf16(o);
  }
}

// ---------------------------------------------------------------------------
// One-shot conversion kernels (outside all hot loops)
// ---------------------------------------------------------------------------
__global__ void convert_bf16_k(const float* __restrict__ X,
                               __bf16* __restrict__ Y, long long n)
{
  long long i = ((long long)blockIdx.x * blockDim.x + threadIdx.x) * 8;
  if (i >= n) return;
  const float4* s = (const float4*)(X + i);
  float4 a = s[0], bq = s[1];
  __bf16 o[8] = { to_bf16(a.x),  to_bf16(a.y),  to_bf16(a.z),  to_bf16(a.w),
                  to_bf16(bq.x), to_bf16(bq.y), to_bf16(bq.z), to_bf16(bq.w) };
  *(uint4*)(Y + i) = *(const uint4*)o;
}

// W[K,N] f32 -> Wt[N,K] bf16 (LDS tile transpose, K-guard for K=1200)
__global__ void transpose_cvt_k(const float* __restrict__ W,
                                __bf16* __restrict__ Wt, int K, int N)
{
  __shared__ float t[32][33];
  const int tx = threadIdx.x, ty = threadIdx.y;
  const int n0 = blockIdx.x * 32, k0 = blockIdx.y * 32;
#pragma unroll
  for (int j = 0; j < 32; j += 8) {
    int k = k0 + ty + j;
    t[ty + j][tx] = (k < K) ? W[(size_t)k * N + n0 + tx] : 0.0f;
  }
  __syncthreads();
#pragma unroll
  for (int j = 0; j < 32; j += 8) {
    int k = k0 + tx;
    if (k < K) Wt[(size_t)(n0 + ty + j) * K + k] = to_bf16(t[tx][ty + j]);
  }
}

__global__ void cbnorm_k(const float* __restrict__ cb, float* __restrict__ cbn,
                         int ncodes)
{
  int n = blockIdx.x * blockDim.x + threadIdx.x;
  if (n >= ncodes) return;
  const float* r = cb + (size_t)n * EMB;
  float s = 0.f;
  for (int k = 0; k < EMB; ++k) s += r[k] * r[k];
  cbn[n] = s;
}

__global__ void init_keys_k(unsigned long long* __restrict__ keys, int m)
{
  int i = blockIdx.x * blockDim.x + threadIdx.x;
  if (i < m) keys[i] = ~0ull;
}

__global__ void extract_idx_k(const unsigned long long* __restrict__ keys,
                              int* __restrict__ out, int m)
{
  int i = blockIdx.x * blockDim.x + threadIdx.x;
  if (i < m) out[i] = (int)(keys[i] & 0xffffffffull);
}

// ---------------------------------------------------------------------------
// Fused WMMA distance + argmin over the codebook.
//   d2(m,n) = ||cb_n||^2 - 2 z_m.cb_n  (row term dropped: argmin-invariant)
// key = (ordered(d2)<<32)|n  -> atomicMin; ties pick smallest n (== argmin).
// ---------------------------------------------------------------------------
__global__ __launch_bounds__(256)
void argmin_codes(const __bf16* __restrict__ Zb, const __bf16* __restrict__ CBb,
                  const float* __restrict__ cbn,
                  unsigned long long* __restrict__ gkeys, int codes_per_split)
{
  __shared__ __bf16 As[128 * 40];
  __shared__ __bf16 Bs[64 * 40];
  __shared__ unsigned long long bestKey[128];

  const int tid  = threadIdx.x;
  const int lane = tid & 31;
  const int wave = tid >> 5;
  const int wm   = (wave >> 1) * 32;
  const int wn   = (wave & 1) * 32;
  const int hi16 = lane >> 4;
  const int mr   = lane & 15;
  const int kb   = hi16 * 8;
  const int m0   = blockIdx.y * 128;
  const int cstart = blockIdx.x * codes_per_split;

  if (tid < 128) bestKey[tid] = ~0ull;
  __syncthreads();

  for (int n0 = cstart; n0 < cstart + codes_per_split; n0 += 64) {
    v8f c[2][2] = {};
    for (int k0 = 0; k0 < EMB; k0 += 32) {
#if USE_TDM
      if (wave == 0)
        tdm_load_2d((unsigned)(uintptr_t)&As[0], Zb + (size_t)m0 * EMB + k0,
                    128u, 32u, (unsigned)(EMB - k0), (unsigned)EMB);
      else if (wave == 1)
        tdm_load_2d((unsigned)(uintptr_t)&Bs[0], CBb + (size_t)n0 * EMB + k0,
                    64u, 32u, (unsigned)(EMB - k0), (unsigned)EMB);
      if (wave < 2) __builtin_amdgcn_s_wait_tensorcnt(0);
#else
      {
        const int r = tid >> 1, cg = (tid & 1) * 16;
        const __bf16* src = Zb + (size_t)(m0 + r) * EMB + k0 + cg;
        *(uint4*)&As[r * 40 + cg]     = ((const uint4*)src)[0];
        *(uint4*)&As[r * 40 + cg + 8] = ((const uint4*)src)[1];
      }
      {
        const int n = tid >> 2, kq = (tid & 3) * 8;
        const __bf16* src = CBb + (size_t)(n0 + n) * EMB + k0 + kq;
        *(uint4*)&Bs[n * 40 + kq] = *(const uint4*)src;
      }
#endif
      __syncthreads();

      FragBf a[2], b[2];
#pragma unroll
      for (int i = 0; i < 2; ++i) {
        const __bf16* p = &As[(wm + 16 * i + mr) * 40 + kb];
        a[i].q[0] = *(const uint4*)p;
        a[i].q[1] = *(const uint4*)(p + 16);
      }
#pragma unroll
      for (int j = 0; j < 2; ++j) {
        const __bf16* p = &Bs[(wn + 16 * j + mr) * 40 + kb];
        b[j].q[0] = *(const uint4*)p;
        b[j].q[1] = *(const uint4*)(p + 16);
      }
#pragma unroll
      for (int i = 0; i < 2; ++i)
#pragma unroll
        for (int j = 0; j < 2; ++j)
          c[i][j] = __builtin_amdgcn_wmma_f32_16x16x32_bf16(
              false, a[i].v, false, b[j].v, (short)0, c[i][j], false, false);
      __syncthreads();
    }

#pragma unroll
    for (int i = 0; i < 2; ++i) {
      const int lrow = wm + 16 * i + hi16 * 8;
#pragma unroll
      for (int r = 0; r < 8; ++r) {
        unsigned long long key = ~0ull;
#pragma unroll
        for (int j = 0; j < 2; ++j) {
          const int col = n0 + wn + 16 * j + mr;
          const float d2 = cbn[col] - 2.0f * c[i][j][r];
          unsigned long long k2 =
              ((unsigned long long)ford(d2) << 32) | (unsigned)col;
          if (k2 < key) key = k2;
        }
#pragma unroll
        for (int mshift = 1; mshift <= 8; mshift <<= 1) {
          unsigned long long o = __shfl_xor(key, mshift, 32);
          if (o < key) key = o;
        }
        if (mr == 0) atomicMin(&bestKey[lrow + r], key);
      }
    }
    __syncthreads();
  }

  if (tid < 128) atomicMin(&gkeys[m0 + tid], bestKey[tid]);
}

// ---------------------------------------------------------------------------
// Host-side orchestration
// ---------------------------------------------------------------------------
struct EncParams {
  const float *w_in, *b_in, *g_in, *be_in, *w_out, *b_out;
  const float *w1[2], *b1[2], *w2[2], *b2[2], *g[2], *bln[2];
  int nres;
};

// JAX tree-flatten order (dict keys sorted):
//   b_in, b_out, be_in, g_in, res[ b, b1, b2, g, w1, w2 ]..., w_in, w_out
static EncParams get_enc(void* const* d_in, int base, int nres) {
  EncParams p; p.nres = nres;
  int i = base;
  p.b_in  = (const float*)d_in[i++];
  p.b_out = (const float*)d_in[i++];
  p.be_in = (const float*)d_in[i++];
  p.g_in  = (const float*)d_in[i++];
  for (int r = 0; r < nres; ++r) {
    p.bln[r] = (const float*)d_in[i++];
    p.b1[r]  = (const float*)d_in[i++];
    p.b2[r]  = (const float*)d_in[i++];
    p.g[r]   = (const float*)d_in[i++];
    p.w1[r]  = (const float*)d_in[i++];
    p.w2[r]  = (const float*)d_in[i++];
  }
  p.w_in  = (const float*)d_in[i++];
  p.w_out = (const float*)d_in[i++];
  return p;
}

struct Ws {
  __bf16 *xbf, *actb, *h1b, *zb, *cbb, *wt;
  float  *actf, *h2f, *cbn;
  unsigned long long* keys;
};

static void run_encoder(const float* x, int din, int hid, const EncParams& p,
                        const float* cb, int ncodes, int* out_idx,
                        const Ws& w, hipStream_t stream)
{
  const int M = M_TOK;
  dim3 blk(256);

  // input -> bf16
  convert_bf16_k<<<(int)((size_t)M * din / 8 / 256), 256, 0, stream>>>(
      x, w.xbf, (long long)M * din);

  // h0 = x @ w_in + b_in  (f32 out, LN consumes)
  transpose_cvt_k<<<dim3(hid / 32, (din + 31) / 32), dim3(32, 8), 0, stream>>>(
      p.w_in, w.wt, din, hid);
  gemm_bf16t<0, true, false><<<dim3(hid / 128, M / 128), blk, 0, stream>>>(
      w.xbf, w.wt, p.b_in, nullptr, w.h2f, nullptr, M, hid, din);

  // act = gelu(LN(h0))  (f32 + bf16)
  if (hid == 1024)
    layernorm_k<1024, true><<<M, 256, 0, stream>>>(w.h2f, p.g_in, p.be_in, w.actf, w.actb);
  else
    layernorm_k<512, true><<<M, 256, 0, stream>>>(w.h2f, p.g_in, p.be_in, w.actf, w.actb);

  for (int r = 0; r < p.nres; ++r) {
    // h1 = gelu(act @ w1 + b1)  (bf16 out, next GEMM consumes)
    transpose_cvt_k<<<dim3(2 * hid / 32, (hid + 31) / 32), dim3(32, 8), 0, stream>>>(
        p.w1[r], w.wt, hid, 2 * hid);
    gemm_bf16t<1, false, true><<<dim3(2 * hid / 128, M / 128), blk, 0, stream>>>(
        w.actb, w.wt, p.b1[r], nullptr, nullptr, w.h1b, M, 2 * hid, hid);
    // h2 = act + h1 @ w2 + b2  (f32 out, LN consumes; residual from actf)
    transpose_cvt_k<<<dim3(hid / 32, (2 * hid + 31) / 32), dim3(32, 8), 0, stream>>>(
        p.w2[r], w.wt, 2 * hid, hid);
    gemm_bf16t<2, true, false><<<dim3(hid / 128, M / 128), blk, 0, stream>>>(
        w.h1b, w.wt, p.b2[r], w.actf, w.h2f, nullptr, M, hid, 2 * hid);
    // act = LN(h2)
    if (hid == 1024)
      layernorm_k<1024, false><<<M, 256, 0, stream>>>(w.h2f, p.g[r], p.bln[r], w.actf, w.actb);
    else
      layernorm_k<512, false><<<M, 256, 0, stream>>>(w.h2f, p.g[r], p.bln[r], w.actf, w.actb);
  }

  // z = act @ w_out + b_out  (bf16 out, argmin consumes)
  transpose_cvt_k<<<dim3(EMB / 32, (hid + 31) / 32), dim3(32, 8), 0, stream>>>(
      p.w_out, w.wt, hid, EMB);
  gemm_bf16t<0, false, true><<<dim3(EMB / 128, M / 128), blk, 0, stream>>>(
      w.actb, w.wt, p.b_out, nullptr, nullptr, w.zb, M, EMB, hid);

  // nearest codebook index
  convert_bf16_k<<<(int)((size_t)ncodes * EMB / 8 / 256), 256, 0, stream>>>(
      cb, w.cbb, (long long)ncodes * EMB);
  cbnorm_k<<<(ncodes + 255) / 256, 256, 0, stream>>>(cb, w.cbn, ncodes);
  init_keys_k<<<(M + 255) / 256, 256, 0, stream>>>(w.keys, M);
  const int nsplit = 4;
  argmin_codes<<<dim3(nsplit, M / 128), blk, 0, stream>>>(
      w.zb, w.cbb, w.cbn, w.keys, ncodes / nsplit);
  extract_idx_k<<<(M + 255) / 256, 256, 0, stream>>>(w.keys, out_idx, M);
}

extern "C" void kernel_launch(void* const* d_in, const int* in_sizes, int n_in,
                              void* d_out, int out_size, void* d_ws, size_t ws_size,
                              hipStream_t stream)
{
  (void)in_sizes; (void)n_in; (void)out_size; (void)ws_size;

  const float* pose   = (const float*)d_in[0];
  const float* motion = (const float*)d_in[1];
  const float* dyn    = (const float*)d_in[2];
  EncParams pp = get_enc(d_in, 3, 2);    // pose:   hid=1024, 2 res blocks
  EncParams mp = get_enc(d_in, 21, 1);   // motion: hid=1024, 1 res block
  EncParams dp = get_enc(d_in, 33, 1);   // dyn:    hid=512,  1 res block
  const float* pose_cb   = (const float*)d_in[45];
  const float* motion_cb = (const float*)d_in[46];
  const float* dyn_cb    = (const float*)d_in[47];
  int* out = (int*)d_out;

  const size_t M = M_TOK;
  char* base = (char*)d_ws;
  size_t off = 0;
  auto alloc = [&](size_t bytes) -> void* {
    void* pch = base + off;
    off = (off + bytes + 255) & ~(size_t)255;
    return pch;
  };
  Ws w;
  w.xbf  = (__bf16*)alloc(M * 1200 * 2);
  w.actf = (float*) alloc(M * 1024 * 4);
  w.actb = (__bf16*)alloc(M * 1024 * 2);
  w.h1b  = (__bf16*)alloc(M * 2048 * 2);
  w.h2f  = (float*) alloc(M * 1024 * 4);
  w.zb   = (__bf16*)alloc(M * EMB * 2);
  w.cbb  = (__bf16*)alloc((size_t)4096 * EMB * 2);
  w.wt   = (__bf16*)alloc((size_t)2048 * 1024 * 2);
  w.keys = (unsigned long long*)alloc(M * 8);
  w.cbn  = (float*)alloc(4096 * 4);

  run_encoder(pose,   1200, 1024, pp, pose_cb,   4096, out,                 w, stream);
  run_encoder(motion, 1200, 1024, mp, motion_cb, 4096, out + M,             w, stream);
  run_encoder(dyn,    1200,  512, dp, dyn_cb,    2048, out + 2 * M,         w, stream);
}
